// SCAM_57294863729467
// MI455X (gfx1250) — matrix operand order repaired and checked by reading, not compiled, not run on hardware
//
#include <hip/hip_runtime.h>
#include <hip/hip_bf16.h>

// ---------------------------------------------------------------------------
// SCAM (stereo cross-attention) for MI455X / gfx1250.
// All matmuls use v_wmma_f32_16x16x32_f16 (wave32). Fragment layouts used here
// (contiguous k order in A and B, not the ISA table order):
//   A (16x32 f16): lane m=lane&15, half g=lane>>4; VGPR v holds K pair at
//                  kb = 16*g + 2*v
//   B (32x16 f16): lane n=lane&15; VGPR v holds K pair at kb = 16*g + 2*v
//   C/D (16x16 f32): VGPR r -> row m = r + 8*g, col n = lane&15
// Tile staging uses GLOBAL_LOAD_ASYNC_TO_LDS_B128 (ASYNCcnt) when available.
// ---------------------------------------------------------------------------

typedef __attribute__((ext_vector_type(16))) _Float16 v16h;
typedef __attribute__((ext_vector_type(8)))  float    v8f;
typedef __attribute__((ext_vector_type(4)))  int      v4i;
typedef unsigned int u32;

#define BDIM   2
#define CCH    192
#define HH     128
#define WW     128
#define HWSZ   (HH*WW)          // 16384
#define PIX    (BDIM*HWSZ)      // 32768
#define NHEAD  6
#define HDIM   32
#define WSZ    16
#define OWSZ   24
#define NWIN   128              // B * 8 * 8 windows per side
#define RPN    1521             // (16+24-1)^2
#define SCALE_F 0.17677669529663689f   // 32^-0.5

#if defined(__has_builtin)
#if __has_builtin(__builtin_amdgcn_global_load_async_to_lds_b128) && \
    __has_builtin(__builtin_amdgcn_s_wait_asynccnt)
#define HAVE_ASYNC_COPY 1
#endif
#endif
#ifndef HAVE_ASYNC_COPY
#define HAVE_ASYNC_COPY 0
#endif

// Pointee types for the async-to-LDS builtin: 4xint vector qualified with the
// global (AS1) / LDS (AS3) address spaces, per the builtin's signature.
typedef __attribute__((address_space(1))) v4i as1_v4i;
typedef __attribute__((address_space(3))) v4i as3_v4i;

// Copy 16 bytes global -> LDS. Async path: data moves directly into LDS
// (no VGPR round trip), tracked by ASYNCcnt.
static __device__ __forceinline__ void copy16_g2l(const void* g, void* l) {
#if HAVE_ASYNC_COPY
  __builtin_amdgcn_global_load_async_to_lds_b128((as1_v4i*)g, (as3_v4i*)l, 0, 0);
#else
  const u32* s = (const u32*)g;
  u32 r0 = s[0], r1 = s[1], r2 = s[2], r3 = s[3];
  u32* d = (u32*)l;
  d[0] = r0; d[1] = r1; d[2] = r2; d[3] = r3;
#endif
}
static __device__ __forceinline__ void copy_join() {
#if HAVE_ASYNC_COPY
  __builtin_amdgcn_s_wait_asynccnt(0);
#endif
}

static __device__ __forceinline__ v8f wmma32(v16h a, v16h b, v8f c) {
  return __builtin_amdgcn_wmma_f32_16x16x32_f16(false, a, false, b,
                                                (short)0, c, false, false);
}

static __device__ __forceinline__ v8f zero8() {
  v8f z = {0.f,0.f,0.f,0.f,0.f,0.f,0.f,0.f};
  return z;
}

// A operand fragment: src is M x K row-major tile origin (M>=16 rows).
static __device__ __forceinline__ v16h load_a_frag(const _Float16* src, int stride) {
  const int lane = threadIdx.x & 31;
  const int g = lane >> 4, m = lane & 15;
  v16h a;
#pragma unroll
  for (int v = 0; v < 8; ++v) {
    const int kb = (g << 4) + (v << 1);
    a[2*v]   = src[m * stride + kb];
    a[2*v+1] = src[m * stride + kb + 1];
  }
  return a;
}

// B operand fragment from K x N row-major storage (k rows, n cols).
static __device__ __forceinline__ v16h load_b_frag(const _Float16* src, int stride) {
  const int lane = threadIdx.x & 31;
  const int g = lane >> 4, n = lane & 15;
  v16h b;
#pragma unroll
  for (int v = 0; v < 8; ++v) {
    const int kb = (g << 4) + (v << 1);
    b[2*v]   = src[kb * stride + n];
    b[2*v+1] = src[(kb + 1) * stride + n];
  }
  return b;
}

// B operand fragment from N x K row-major storage (n rows, k cols; K pairs contiguous).
static __device__ __forceinline__ v16h load_bT_frag(const _Float16* src, int stride) {
  const int lane = threadIdx.x & 31;
  const int g = lane >> 4, n = lane & 15;
  v16h b;
#pragma unroll
  for (int v = 0; v < 8; ++v) {
    const int kb = (g << 4) + (v << 1);
    b[2*v]   = src[n * stride + kb];
    b[2*v+1] = src[n * stride + kb + 1];
  }
  return b;
}

// --------------------------------------------------------------------------
// f32 -> f16 weight conversion
// --------------------------------------------------------------------------
__global__ void cvt_kernel(const float* __restrict__ src, _Float16* __restrict__ dst, int n) {
  int i = blockIdx.x * 256 + threadIdx.x;
  if (i < n) dst[i] = (_Float16)src[i];
}

// --------------------------------------------------------------------------
// LN over channels of NCHW input -> NHWC f32 (residual) + NHWC f16 (GEMM A).
// 64 pixels/block, LDS transpose keeps global reads and writes coalesced.
// --------------------------------------------------------------------------
__global__ void ln1_kernel(const float* __restrict__ x,
                           const float* __restrict__ gw, const float* __restrict__ bw,
                           float* __restrict__ nf, _Float16* __restrict__ nh) {
  __shared__ float tile[64][196];
  const int t = threadIdx.x;                 // 256 threads
  const int pix0 = blockIdx.x * 64;
  const int b = pix0 >> 14;
  const int hw0 = pix0 & (HWSZ - 1);
#pragma unroll 4
  for (int cb = 0; cb < 48; ++cb) {
    const int c = cb * 4 + (t >> 6);
    const int p = t & 63;
    tile[p][c] = x[((size_t)b * CCH + c) * HWSZ + hw0 + p];
  }
  __syncthreads();
  const int wave = t >> 5, lane = t & 31;
  for (int pp = 0; pp < 8; ++pp) {
    const int p = wave * 8 + pp;
    float s = 0.f, s2 = 0.f;
#pragma unroll
    for (int j = 0; j < 6; ++j) {
      const float v = tile[p][lane + 32 * j];
      s += v; s2 += v * v;
    }
#pragma unroll
    for (int m = 1; m < 32; m <<= 1) { s += __shfl_xor(s, m); s2 += __shfl_xor(s2, m); }
    const float mu = s * (1.f / CCH);
    const float rstd = rsqrtf(s2 * (1.f / CCH) - mu * mu + 1e-6f);
    const size_t ob = ((size_t)pix0 + p) * CCH;
#pragma unroll
    for (int j = 0; j < 6; ++j) {
      const int c = lane + 32 * j;
      const float v = (tile[p][c] - mu) * rstd * gw[c] + bw[c];
      nf[ob + c] = v;
      nh[ob + c] = (_Float16)v;
    }
  }
}

// --------------------------------------------------------------------------
// LN over channels of NHWC f16 input -> NHWC f16 (wave per pixel, coalesced).
// --------------------------------------------------------------------------
__global__ void ln2_kernel(const _Float16* __restrict__ xin,
                           const float* __restrict__ gw, const float* __restrict__ bw,
                           _Float16* __restrict__ yout) {
  const int lane = threadIdx.x & 31;
  const int wv = blockIdx.x * (blockDim.x >> 5) + (threadIdx.x >> 5);
  const _Float16* src = xin + (size_t)wv * CCH;
  float v[6], s = 0.f, s2 = 0.f;
#pragma unroll
  for (int j = 0; j < 6; ++j) {
    v[j] = (float)src[lane + 32 * j];
    s += v[j]; s2 += v[j] * v[j];
  }
#pragma unroll
  for (int m = 1; m < 32; m <<= 1) { s += __shfl_xor(s, m); s2 += __shfl_xor(s2, m); }
  const float mu = s * (1.f / CCH);
  const float rstd = rsqrtf(s2 * (1.f / CCH) - mu * mu + 1e-6f);
  _Float16* dst = yout + (size_t)wv * CCH;
#pragma unroll
  for (int j = 0; j < 6; ++j) {
    const int c = lane + 32 * j;
    dst[c] = (_Float16)((v[j] - mu) * rstd * gw[c] + bw[c]);
  }
}

// --------------------------------------------------------------------------
// Generic Y[m][n] = (sum_k A[m][k]*W[n][k] + bias[n]) * scale  (f16 in/out).
// 128 threads = 4 waves, 64x64 block tile, async-LDS-staged, f16 WMMA.
// out_mode 1 remaps row m=(b,h,w) -> (b,w,h) for the Q projections.
// --------------------------------------------------------------------------
__global__ void gemm_kernel(const _Float16* __restrict__ A,
                            const _Float16* __restrict__ Wt,   // N x K
                            const float* __restrict__ bias,
                            _Float16* __restrict__ Y,
                            int M, int N, int K, int out_mode, float scale) {
  __shared__ __align__(16) _Float16 As[64 * 40];
  __shared__ __align__(16) _Float16 Bs[64 * 40];
  const int t = threadIdx.x;
  const int lane = t & 31, wave = t >> 5;
  const int wr = wave >> 1, wc = wave & 1;
  const int g = lane >> 4, nn = lane & 15;
  const int m0 = blockIdx.x * 64, n0 = blockIdx.y * 64;
  v8f acc[2][2] = {};
  for (int k0 = 0; k0 < K; k0 += 32) {
    // 64x32 f16 tiles: 256 x 16B segments each, async direct-to-LDS.
#pragma unroll
    for (int j = 0; j < 2; ++j) {
      const int e = t + j * 128;            // 16B segment index
      const int row = e >> 2, sp = (e & 3) << 3;
      copy16_g2l(&A [(size_t)(m0 + row) * K + k0 + sp], &As[row * 40 + sp]);
      copy16_g2l(&Wt[(size_t)(n0 + row) * K + k0 + sp], &Bs[row * 40 + sp]);
    }
    if (k0 + 32 < K)                        // gfx1250 global_prefetch_b8
      __builtin_prefetch(&A[(size_t)(m0 + (t & 63)) * K + k0 + 32], 0, 0);
    copy_join();
    __syncthreads();
    v16h af[2], bf[2];
    af[0] = load_a_frag (As + (wr * 32     ) * 40, 40);
    af[1] = load_a_frag (As + (wr * 32 + 16) * 40, 40);
    bf[0] = load_bT_frag(Bs + (wc * 32     ) * 40, 40);
    bf[1] = load_bT_frag(Bs + (wc * 32 + 16) * 40, 40);
#pragma unroll
    for (int tr = 0; tr < 2; ++tr)
#pragma unroll
      for (int tc = 0; tc < 2; ++tc)
        acc[tr][tc] = wmma32(af[tr], bf[tc], acc[tr][tc]);
    __syncthreads();
  }
#pragma unroll
  for (int tr = 0; tr < 2; ++tr)
#pragma unroll
    for (int tc = 0; tc < 2; ++tc)
#pragma unroll
      for (int r = 0; r < 8; ++r) {
        const int gm = m0 + wr * 32 + tr * 16 + r + 8 * g;
        const int gn = n0 + wc * 32 + tc * 16 + nn;
        const float v = (acc[tr][tc][r] + bias[gn]) * scale;
        size_t row;
        if (out_mode == 0) row = (size_t)gm;
        else {
          const int b = gm >> 14, hw = gm & (HWSZ - 1);
          const int h = hw >> 7, w = hw & 127;
          row = (size_t)b * HWSZ + (size_t)w * HH + h;
        }
        Y[row * (size_t)N + gn] = (_Float16)v;
      }
}

// --------------------------------------------------------------------------
// Overlapping-window attention (OCA), one window per block, flash softmax.
// q: 256x32, k/v: 576x32 per head async-staged in LDS (stride 40 halves =
// 80 B rows -> 16B-aligned for B128 and bank-conflict-free fragment reads);
// rel-pos bias computed on the fly from an LDS-staged rpb column.
// --------------------------------------------------------------------------
__device__ __forceinline__ float oca_bias(int qr, int qc, int j, const float* rpbs) {
  const int r2 = (j * 2731) >> 16;            // j / 24
  const int c2 = j - r2 * 24;
  int idx = (r2 - qr - 7) * 39 + (c2 - qc - 7);
  if (idx < 0) idx += RPN;
  return rpbs[idx];
}

__global__ void oca_kernel(const _Float16* __restrict__ qkv,   // (B,H,W,576) f16
                           const float* __restrict__ rpb,      // (1521,6) f32
                           _Float16* __restrict__ ao) {        // (B,H,W,192) f16
  extern __shared__ __align__(16) _Float16 oca_smem[];
  _Float16* qs = oca_smem;                    // 256 x 40
  _Float16* ks = qs + 256 * 40;               // 576 x 40
  _Float16* vs = ks + 576 * 40;               // 576 x 40
  _Float16* pb = vs + 576 * 40;               // 8 waves x 16 x 40
  float* rpbs  = (float*)(pb + 8 * 16 * 40);  // 1521

  const int win = blockIdx.x;
  const int b = win >> 6, wh = (win >> 3) & 7, ww = win & 7;
  const int t = threadIdx.x, lane = t & 31, wave = t >> 5;
  const int g = lane >> 4, nn = lane & 15;

  for (int head = 0; head < NHEAD; ++head) {
    __syncthreads();
    for (int i = t; i < RPN; i += 256) rpbs[i] = rpb[i * NHEAD + head];
    {   // q tile: one pixel per thread, 32 f16 = 4 x 16B async transfers
      const int p = t, r = p >> 4, c = p & 15;
      const int h = wh * WSZ + r, w = ww * WSZ + c;
      const _Float16* s = qkv + (((size_t)(b * HH + h)) * WW + w) * 576 + head * HDIM;
#pragma unroll
      for (int j = 0; j < 4; ++j) copy16_g2l(s + j * 8, qs + p * 40 + j * 8);
    }
    for (int p = t; p < 576; p += 256) {      // k/v tiles with zero padding
      const int r = (p * 2731) >> 16;
      const int c = p - r * 24;
      const int h = wh * WSZ - 4 + r, w = ww * WSZ - 4 + c;
      if (h >= 0 && h < HH && w >= 0 && w < WW) {
        const _Float16* base = qkv + (((size_t)(b * HH + h)) * WW + w) * 576;
#pragma unroll
        for (int j = 0; j < 4; ++j) {
          copy16_g2l(base + CCH     + head * HDIM + j * 8, ks + p * 40 + j * 8);
          copy16_g2l(base + 2 * CCH + head * HDIM + j * 8, vs + p * 40 + j * 8);
        }
      } else {
        u32* dk = (u32*)(ks + p * 40);
        u32* dv = (u32*)(vs + p * 40);
#pragma unroll
        for (int j = 0; j < 16; ++j) { dk[j] = 0u; dv[j] = 0u; }
      }
    }
    copy_join();
    __syncthreads();

    for (int rt2 = 0; rt2 < 2; ++rt2) {
      const int rt = wave * 2 + rt2;
      const v16h qf = load_a_frag(qs + rt * 16 * 40, 40);
      _Float16* myp = pb + wave * 16 * 40;
      float m8[8], l8[8];
      v8f acc0 = zero8(), acc1 = zero8();
#pragma unroll
      for (int r = 0; r < 8; ++r) { m8[r] = -1e30f; l8[r] = 0.f; }

      for (int pr = 0; pr < 18; ++pr) {       // 2 key tiles (K=32) per step
        const v16h kf0 = load_bT_frag(ks + (pr * 32     ) * 40, 40);
        const v16h kf1 = load_bT_frag(ks + (pr * 32 + 16) * 40, 40);
        v8f ta = wmma32(qf, kf0, zero8());
        v8f tb = wmma32(qf, kf1, zero8());
        const int kbase = pr * 32;
#pragma unroll
        for (int r = 0; r < 8; ++r) {
          const int qi = rt * 16 + r + 8 * g;
          const int qr = qi >> 4, qc = qi & 15;
          ta[r] = ta[r] * SCALE_F + oca_bias(qr, qc, kbase + nn, rpbs);
          tb[r] = tb[r] * SCALE_F + oca_bias(qr, qc, kbase + 16 + nn, rpbs);
          // online softmax over this 16x32 strip (row spans 16 lanes of a half)
          float x = fmaxf(ta[r], tb[r]);
          x = fmaxf(x, __shfl_xor(x, 1));  x = fmaxf(x, __shfl_xor(x, 2));
          x = fmaxf(x, __shfl_xor(x, 4));  x = fmaxf(x, __shfl_xor(x, 8));
          const float nm = fmaxf(m8[r], x);
          const float sc = __expf(m8[r] - nm);
          const float pa = __expf(ta[r] - nm);
          const float pv = __expf(tb[r] - nm);
          float ss = pa + pv;
          ss += __shfl_xor(ss, 1); ss += __shfl_xor(ss, 2);
          ss += __shfl_xor(ss, 4); ss += __shfl_xor(ss, 8);
          l8[r] = l8[r] * sc + ss;
          m8[r] = nm;
          acc0[r] *= sc; acc1[r] *= sc;
          const int m = r + 8 * g;
          myp[m * 40 + nn]      = (_Float16)pa;   // C/D layout -> f16 A tile
          myp[m * 40 + 16 + nn] = (_Float16)pv;
        }
        asm volatile("s_wait_dscnt 0x0" ::: "memory");   // LDS RAW across lanes
        const v16h af  = load_a_frag(myp, 40);
        const v16h vf0 = load_b_frag(vs + (size_t)(pr * 32) * 40,      40);
        const v16h vf1 = load_b_frag(vs + (size_t)(pr * 32) * 40 + 16, 40);
        acc0 = wmma32(af, vf0, acc0);
        acc1 = wmma32(af, vf1, acc1);
        asm volatile("" ::: "memory");
      }
#pragma unroll
      for (int r = 0; r < 8; ++r) {
        const float inv = 1.f / l8[r];
        const int qi = rt * 16 + r + 8 * g;
        const int h = wh * WSZ + (qi >> 4), w = ww * WSZ + (qi & 15);
        _Float16* dst = ao + (((size_t)(b * HH + h)) * WW + w) * CCH + head * HDIM;
        dst[nn]      = (_Float16)(acc0[r] * inv);
        dst[16 + nn] = (_Float16)(acc1[r] * inv);
      }
    }
  }
}

// --------------------------------------------------------------------------
// Bidirectional stereo cross attention, one (b,t) slice per block.
// att (128x128 f32) lives in LDS; both softmax directions + both P x V
// GEMMs run from LDS. 200 KB dynamic LDS (WGP has 320 KB).
// --------------------------------------------------------------------------
__device__ __forceinline__ void pv_gemm(const _Float16* P, const _Float16* V,
                                        _Float16* out) {
  const int t = threadIdx.x, lane = t & 31, wave = t >> 5;
  const int g = lane >> 4, nn = lane & 15;
  v16h af[4];
#pragma unroll
  for (int kc = 0; kc < 4; ++kc)
    af[kc] = load_a_frag(P + (size_t)(wave * 16) * 136 + kc * 32, 136);
  for (int ct = 0; ct < 12; ++ct) {
    v8f acc = zero8();
#pragma unroll
    for (int kc = 0; kc < 4; ++kc) {
      const v16h bf = load_b_frag(V + (size_t)(kc * 32) * 200 + ct * 16, 200);
      acc = wmma32(af[kc], bf, acc);
    }
#pragma unroll
    for (int r = 0; r < 8; ++r)
      out[(size_t)(wave * 16 + r + 8 * g) * CCH + ct * 16 + nn] = (_Float16)acc[r];
  }
}

// Stage two contiguous 128x192 f16 slices into padded 128x200 LDS buffers.
static __device__ __forceinline__ void stage_slices(const _Float16* s0,
                                                    const _Float16* s1,
                                                    _Float16* d0, _Float16* d1) {
  const int t = threadIdx.x;
  for (int e = t; e < 3072; e += 256) {        // 16B segments: 128 rows x 24
    const int row = e / 24, sp = (e - row * 24) * 8;
    copy16_g2l(s0 + (size_t)row * CCH + sp, d0 + row * 200 + sp);
    copy16_g2l(s1 + (size_t)row * CCH + sp, d1 + row * 200 + sp);
  }
  copy_join();
}

__global__ void cross_kernel(const _Float16* __restrict__ Ql,  // (B,W,H,C) (scale folded)
                             const _Float16* __restrict__ Qr,  // (B,W,H,C)
                             const _Float16* __restrict__ Vl,  // (B,H,W,C)
                             const _Float16* __restrict__ Vr,  // (B,H,W,C)
                             _Float16* __restrict__ Fr2l,      // (B,H,W,C)
                             _Float16* __restrict__ Fl2r) {    // (B,H,W,C)
  extern __shared__ __align__(16) unsigned char cross_smem[];
  float*    att  = (float*)cross_smem;                        // 128 x 132
  _Float16* bufA = (_Float16*)(cross_smem + 128 * 132 * 4);   // 128 x 200
  _Float16* bufB = bufA + 128 * 200;                          // 128 x 200
  _Float16* pbuf = bufB + 128 * 200;                          // 128 x 136

  const int bt = blockIdx.x;                 // b*128 + t
  const size_t base = (size_t)bt * HH * CCH;
  const int t = threadIdx.x, lane = t & 31, wave = t >> 5;
  const int g = lane >> 4, nn = lane & 15;

  stage_slices(Ql + base, Qr + base, bufA, bufB);
  __syncthreads();
  {   // att = Ql * Qr^T  (each wave: one 16-row block x 128 cols, K=192)
    v16h af[6];
#pragma unroll
    for (int kc = 0; kc < 6; ++kc)
      af[kc] = load_a_frag(bufA + (size_t)(wave * 16) * 200 + kc * 32, 200);
    for (int ct = 0; ct < 8; ++ct) {
      v8f acc = zero8();
#pragma unroll
      for (int kc = 0; kc < 6; ++kc) {
        const v16h bf = load_bT_frag(bufB + (size_t)(ct * 16) * 200 + kc * 32, 200);
        acc = wmma32(af[kc], bf, acc);
      }
#pragma unroll
      for (int r = 0; r < 8; ++r)
        att[(wave * 16 + r + 8 * g) * 132 + ct * 16 + nn] = acc[r];
    }
  }
  __syncthreads();
  stage_slices(Vr + base, Vl + base, bufA, bufB);   // reuse operand buffers
  __syncthreads();
  {   // row softmax -> pbuf (f16)
    const int row = t >> 1, half = t & 1;
    const float* ar = att + row * 132 + half * 64;
    float mx = -1e30f;
    for (int j = 0; j < 64; ++j) mx = fmaxf(mx, ar[j]);
    mx = fmaxf(mx, __shfl_xor(mx, 1));
    float s = 0.f;
    for (int j = 0; j < 64; ++j) s += __expf(ar[j] - mx);
    s += __shfl_xor(s, 1);
    const float inv = 1.f / s;
    _Float16* prow = pbuf + row * 136 + half * 64;
    for (int j = 0; j < 64; ++j) prow[j] = (_Float16)(__expf(ar[j] - mx) * inv);
  }
  __syncthreads();
  pv_gemm(pbuf, bufA, Fr2l + base);          // F_r2l = softmax(att) x Vr
  __syncthreads();
  {   // column softmax -> pbuf (transposed probabilities)
    const int col = t >> 1, half = t & 1;
    float mx = -1e30f;
    for (int i = 0; i < 64; ++i) mx = fmaxf(mx, att[(half * 64 + i) * 132 + col]);
    mx = fmaxf(mx, __shfl_xor(mx, 1));
    float s = 0.f;
    for (int i = 0; i < 64; ++i) s += __expf(att[(half * 64 + i) * 132 + col] - mx);
    s += __shfl_xor(s, 1);
    const float inv = 1.f / s;
    _Float16* prow = pbuf + col * 136 + half * 64;
    for (int i = 0; i < 64; ++i)
      prow[i] = (_Float16)(__expf(att[(half * 64 + i) * 132 + col] - mx) * inv);
  }
  __syncthreads();
  pv_gemm(pbuf, bufB, Fl2r + base);          // F_l2r = softmax(att^T) x Vl
}

// --------------------------------------------------------------------------
// out_nchw[b,c,h,w] = nf_nhwc + F_nhwc * coef[c], LDS transpose for coalescing
// --------------------------------------------------------------------------
__global__ void out_kernel(const float* __restrict__ nf, const _Float16* __restrict__ F,
                           const float* __restrict__ coef, float* __restrict__ out) {
  __shared__ float tile[CCH * 33];
  const int bi = blockIdx.x;
  const int wt = bi & 3, h = (bi >> 2) & 127, b = bi >> 9;
  const int w0 = wt * 32;
  const int t = threadIdx.x;
  const size_t ibase = (((size_t)(b * HH + h)) * WW + w0) * CCH;
  for (int e = t; e < 32 * CCH; e += 256) {
    const int wi = e / CCH, c = e - wi * CCH;
    tile[c * 33 + wi] = nf[ibase + (size_t)wi * CCH + c]
                      + (float)F[ibase + (size_t)wi * CCH + c] * coef[c];
  }
  __syncthreads();
  for (int e = t; e < 32 * CCH; e += 256) {
    const int c = e >> 5, wi = e & 31;
    out[(((size_t)b * CCH + c) * HH + h) * WW + w0 + wi] = tile[c * 33 + wi];
  }
}

// --------------------------------------------------------------------------
extern "C" void kernel_launch(void* const* d_in, const int* in_sizes, int n_in,
                              void* d_out, int out_size, void* d_ws, size_t ws_size,
                              hipStream_t stream) {
  (void)in_sizes; (void)n_in; (void)out_size;
  const float* x_l   = (const float*)d_in[0];
  const float* x_r   = (const float*)d_in[1];
  const float* qkv_w = (const float*)d_in[2];
  const float* qkv_b = (const float*)d_in[3];
  const float* rpb   = (const float*)d_in[4];
  const float* nlg   = (const float*)d_in[5];
  const float* nlb   = (const float*)d_in[6];
  const float* nrg   = (const float*)d_in[7];
  const float* nrb   = (const float*)d_in[8];
  const float* lp1w  = (const float*)d_in[9];
  const float* lp1b  = (const float*)d_in[10];
  const float* rp1w  = (const float*)d_in[11];
  const float* rp1b  = (const float*)d_in[12];
  const float* lp2w  = (const float*)d_in[13];
  const float* lp2b  = (const float*)d_in[14];
  const float* rp2w  = (const float*)d_in[15];
  const float* rp2b  = (const float*)d_in[16];
  const float* beta  = (const float*)d_in[17];
  const float* gamma = (const float*)d_in[18];

  char* ws = (char*)d_ws;
  size_t off = 0;
  auto take = [&](size_t bytes) {
    char* p = ws + off;
    off += (bytes + 255) & ~(size_t)255;
    return p;
  };
  const size_t NF = (size_t)PIX * CCH * sizeof(float);      // 25165824
  const size_t NH = (size_t)PIX * CCH * sizeof(_Float16);   // 12582912

  float*    nf_l = (float*)take(NF);
  float*    nf_r = (float*)take(NF);
  _Float16* nh_l = (_Float16*)take(NH);
  _Float16* nh_r = (_Float16*)take(NH);
  _Float16* wq   = (_Float16*)take(576 * CCH * 2);
  _Float16* wl1  = (_Float16*)take(CCH * CCH * 2);
  _Float16* wr1  = (_Float16*)take(CCH * CCH * 2);
  _Float16* wl2  = (_Float16*)take(CCH * CCH * 2);
  _Float16* wr2  = (_Float16*)take(CCH * CCH * 2);
  _Float16* qkvl = (_Float16*)take((size_t)PIX * 576 * 2);  // reused as a2_l
  _Float16* qkvr = (_Float16*)take((size_t)PIX * 576 * 2);  // reused as a2_r
  _Float16* ao_l = (_Float16*)take(NH);                     // reused as Fr2l
  _Float16* ao_r = (_Float16*)take(NH);                     // reused as Fl2r
  _Float16* Qlb  = (_Float16*)take(NH);
  _Float16* Qrb  = (_Float16*)take(NH);
  _Float16* Vlb  = (_Float16*)take(NH);
  _Float16* Vrb  = (_Float16*)take(NH);
  if (off > ws_size) return;   // workspace too small (~217 MiB required)

  _Float16* a2_l = qkvl;   // LN2 output overwrites dead qkv buffers
  _Float16* a2_r = qkvr;
  _Float16* Fr2l = ao_l;   // cross-attn output overwrites dead attn-out
  _Float16* Fl2r = ao_r;

  // 1) weights -> f16
  cvt_kernel<<<(576 * CCH + 255) / 256, 256, 0, stream>>>(qkv_w, wq, 576 * CCH);
  cvt_kernel<<<(CCH * CCH + 255) / 256, 256, 0, stream>>>(lp1w, wl1, CCH * CCH);
  cvt_kernel<<<(CCH * CCH + 255) / 256, 256, 0, stream>>>(rp1w, wr1, CCH * CCH);
  cvt_kernel<<<(CCH * CCH + 255) / 256, 256, 0, stream>>>(lp2w, wl2, CCH * CCH);
  cvt_kernel<<<(CCH * CCH + 255) / 256, 256, 0, stream>>>(rp2w, wr2, CCH * CCH);

  // 2) LN1
  ln1_kernel<<<PIX / 64, 256, 0, stream>>>(x_l, nlg, nlb, nf_l, nh_l);
  ln1_kernel<<<PIX / 64, 256, 0, stream>>>(x_r, nrg, nrb, nf_r, nh_r);

  // 3) QKV projection GEMMs (32768 x 576 x 192)
  dim3 gq(PIX / 64, 576 / 64);
  gemm_kernel<<<gq, 128, 0, stream>>>(nh_l, wq, qkv_b, qkvl, PIX, 576, CCH, 0, 1.f);
  gemm_kernel<<<gq, 128, 0, stream>>>(nh_r, wq, qkv_b, qkvr, PIX, 576, CCH, 0, 1.f);

  // 4) overlapping-window attention (flash softmax, WMMA)
  const size_t oca_lds = (256 * 40 + 2 * 576 * 40 + 8 * 16 * 40) * 2 + RPN * 4;
  oca_kernel<<<NWIN, 256, oca_lds, stream>>>(qkvl, rpb, ao_l);
  oca_kernel<<<NWIN, 256, oca_lds, stream>>>(qkvr, rpb, ao_r);

  // 5) LN2 (wave per pixel)
  ln2_kernel<<<PIX / 8, 256, 0, stream>>>(ao_l, nlg, nlb, a2_l);
  ln2_kernel<<<PIX / 8, 256, 0, stream>>>(ao_r, nrg, nrb, a2_r);

  // 6) Q (HW-transposed, SCALE folded into left side) and V projections
  dim3 gp(PIX / 64, CCH / 64);
  gemm_kernel<<<gp, 128, 0, stream>>>(a2_l, wl1, lp1b, Qlb, PIX, CCH, CCH, 1, SCALE_F);
  gemm_kernel<<<gp, 128, 0, stream>>>(a2_r, wr1, rp1b, Qrb, PIX, CCH, CCH, 1, 1.f);
  gemm_kernel<<<gp, 128, 0, stream>>>(nh_l, wl2, lp2b, Vlb, PIX, CCH, CCH, 0, 1.f);
  gemm_kernel<<<gp, 128, 0, stream>>>(nh_r, wr2, rp2b, Vrb, PIX, CCH, CCH, 0, 1.f);

  // 7) bidirectional cross attention (200 KB dynamic LDS per workgroup)
  const size_t cross_lds = 128 * 132 * 4 + 2 * 128 * 200 * 2 + 128 * 136 * 2;
  cross_kernel<<<BDIM * HH, 256, cross_lds, stream>>>(Qlb, Qrb, Vlb, Vrb, Fr2l, Fl2r);

  // 8) residual + NCHW transpose
  out_kernel<<<BDIM * HH * 4, 256, 0, stream>>>(nf_l, Fr2l, beta, (float*)d_out);
  out_kernel<<<BDIM * HH * 4, 256, 0, stream>>>(nf_r, Fl2r, gamma,
                                                (float*)d_out + (size_t)BDIM * CCH * HWSZ);
}
